// GPSConvNN_86938728005833
// MI455X (gfx1250) — matrix operand order repaired and checked by reading, not compiled, not run on hardware
//
#include <hip/hip_runtime.h>
#include <hip/hip_bf16.h>

typedef __bf16 bf16;
typedef __attribute__((ext_vector_type(16))) __bf16 v16bf;
typedef __attribute__((ext_vector_type(8)))  __bf16 v8bf;
typedef __attribute__((ext_vector_type(8)))  float  v8f;

#define NNODES 4096
#define NCH    256
#define NHEAD  4
#define NEDGE  131072

// ---------------------------------------------------------------------------
// WMMA helpers (CDNA5 gfx1250: v_wmma_f32_16x16x32_bf16, wave32)
// ---------------------------------------------------------------------------
static __device__ __forceinline__ v8f wmma_bf16_f32(v16bf a, v16bf b, v8f c) {
  // 8 args: (neg_a, A, neg_b, B, c_mod, C, reuse_a, reuse_b)
  return __builtin_amdgcn_wmma_f32_16x16x32_bf16(false, a, false, b, (short)0, c,
                                                 false, false);
}

// A fragment (16x32 bf16) of row-major [M x K]: lane = M (L&15),
// half-group (L>>4) selects K offset +8; elems 0..7 -> K=kb..kb+7, 8..15 -> K=kb+16..kb+23
static __device__ __forceinline__ v16bf load_a_frag(const bf16* __restrict__ A,
                                                    int ld, int row0, int k0, int lane) {
  const bf16* p = A + (size_t)(row0 + (lane & 15)) * ld + k0 + ((lane >> 4) << 3);
  v8bf lo = *(const v8bf*)p;
  v8bf hi = *(const v8bf*)(p + 16);
  v16bf a;
#pragma unroll
  for (int i = 0; i < 8; ++i) { a[i] = lo[i]; a[i + 8] = hi[i]; }
  return a;
}

// B fragment (32x16 bf16): B[k][n] = W[col0+n][k] with W row-major [N x K].
// lane = n (L&15), lanes 16-31 hold K+8; elems 0..7 -> K=kb..kb+7, 8..15 -> K=kb+16..kb+23.
static __device__ __forceinline__ v16bf load_b_frag(const bf16* __restrict__ W,
                                                    int ld, int col0, int k0, int lane) {
  const bf16* p = W + (size_t)(col0 + (lane & 15)) * ld + k0 + ((lane >> 4) << 3);
  v8bf lo = *(const v8bf*)p;
  v8bf hi = *(const v8bf*)(p + 16);
  v16bf b;
#pragma unroll
  for (int i = 0; i < 8; ++i) { b[i] = lo[i]; b[i + 8] = hi[i]; }
  return b;
}

// ---------------------------------------------------------------------------
// Generic GEMM: out[M,N] = act(A[M,K] @ W[N,K]^T + bias + add)
// block = 256 threads = 8 waves (2 row x 4 col); wave tile 32x32 ->
// 4 independent accumulator chains, 4 WMMAs per 4 fragment loads.
// Block tile 64x128.  act: 0 = none, 1 = relu
// ---------------------------------------------------------------------------
__global__ void k_gemm(const bf16* __restrict__ A, const bf16* __restrict__ W,
                       const float* __restrict__ bias, const float* __restrict__ add,
                       float* __restrict__ out, int M, int N, int K, int act) {
  const int lane = threadIdx.x & 31;
  const int w    = threadIdx.x >> 5;
  const int row0 = blockIdx.x * 64  + (w & 1) * 32;
  const int col0 = blockIdx.y * 128 + (w >> 1) * 32;
  v8f acc00 = {}, acc01 = {}, acc10 = {}, acc11 = {};
  for (int k0 = 0; k0 < K; k0 += 32) {
    const v16bf a0 = load_a_frag(A, K, row0,      k0, lane);
    const v16bf a1 = load_a_frag(A, K, row0 + 16, k0, lane);
    const v16bf b0 = load_b_frag(W, K, col0,      k0, lane);
    const v16bf b1 = load_b_frag(W, K, col0 + 16, k0, lane);
    acc00 = wmma_bf16_f32(a0, b0, acc00);
    acc01 = wmma_bf16_f32(a0, b1, acc01);
    acc10 = wmma_bf16_f32(a1, b0, acc10);
    acc11 = wmma_bf16_f32(a1, b1, acc11);
  }
  // D layout: VGPR r -> row = r + 8*(lane>=16); col = lane&15
  const int cl    = lane & 15;
  const int rbase = (lane >> 4) << 3;
  const float bv0 = bias ? bias[col0 + cl]      : 0.0f;
  const float bv1 = bias ? bias[col0 + 16 + cl] : 0.0f;
#pragma unroll
  for (int r = 0; r < 8; ++r) {
    const int ra = row0 + rbase + r;
    const int rb = ra + 16;
    float v00 = acc00[r] + bv0, v01 = acc01[r] + bv1;
    float v10 = acc10[r] + bv0, v11 = acc11[r] + bv1;
    if (add) {
      v00 += add[(size_t)ra * N + col0 + cl];
      v01 += add[(size_t)ra * N + col0 + 16 + cl];
      v10 += add[(size_t)rb * N + col0 + cl];
      v11 += add[(size_t)rb * N + col0 + 16 + cl];
    }
    if (act == 1) {
      v00 = fmaxf(v00, 0.0f); v01 = fmaxf(v01, 0.0f);
      v10 = fmaxf(v10, 0.0f); v11 = fmaxf(v11, 0.0f);
    }
    out[(size_t)ra * N + col0 + cl]      = v00;
    out[(size_t)ra * N + col0 + 16 + cl] = v01;
    out[(size_t)rb * N + col0 + cl]      = v10;
    out[(size_t)rb * N + col0 + 16 + cl] = v11;
  }
}

// ---------------------------------------------------------------------------
// Flash-style multi-head attention. Q,Kh: [H][N][64] bf16 (Q pre-scaled by
// 1/sqrt(64)); Vt: [H][64][N] bf16 (d-major); O: [N][256] f32.
// grid = (N/128, H); block = 256 threads = 8 waves; wave = 16 query rows.
// ---------------------------------------------------------------------------
__global__ void k_attn(const bf16* __restrict__ Q, const bf16* __restrict__ Kh,
                       const bf16* __restrict__ Vt, float* __restrict__ O, int n) {
  __shared__ float ldsP[8][16][33];   // per-wave 16x32 prob tile (+pad)
  const int lane = threadIdx.x & 31;
  const int w    = threadIdx.x >> 5;
  const int h    = blockIdx.y;
  const int q0   = blockIdx.x * 128 + w * 16;

  const bf16* qp = Q  + ((size_t)h * n + q0) * 64;
  const bf16* kp = Kh + (size_t)h * n * 64;
  const bf16* vt = Vt + (size_t)h * 64 * n;

  const v16bf aq0 = load_a_frag(qp, 64, 0, 0,  lane);
  const v16bf aq1 = load_a_frag(qp, 64, 0, 32, lane);

  float runM[8], runL[8];
  v8f o0 = {}, o1 = {}, o2 = {}, o3 = {};
#pragma unroll
  for (int r = 0; r < 8; ++r) { runM[r] = -1e30f; runL[r] = 0.0f; }

  for (int kb = 0; kb < n; kb += 32) {
    // ---- scores S[16 x 32] = Q @ K^T : two 16x16 D frags, K-dim = 64 ----
    v8f s0 = {}, s1 = {};
    {
      v16bf b00 = load_b_frag(kp, 64, kb,      0,  lane);
      v16bf b01 = load_b_frag(kp, 64, kb,      32, lane);
      s0 = wmma_bf16_f32(aq0, b00, s0);
      s0 = wmma_bf16_f32(aq1, b01, s0);
      v16bf b10 = load_b_frag(kp, 64, kb + 16, 0,  lane);
      v16bf b11 = load_b_frag(kp, 64, kb + 16, 32, lane);
      s1 = wmma_bf16_f32(aq0, b10, s1);
      s1 = wmma_bf16_f32(aq1, b11, s1);
    }
    // ---- streaming softmax (rows live in 16-lane halves of D layout) ----
    float alpha[8];
#pragma unroll
    for (int r = 0; r < 8; ++r) {
      float mx = fmaxf(s0[r], s1[r]);
#pragma unroll
      for (int m = 8; m >= 1; m >>= 1) mx = fmaxf(mx, __shfl_xor(mx, m, 32));
      const float nm = fmaxf(runM[r], mx);
      alpha[r] = __expf(runM[r] - nm);
      runM[r]  = nm;
    }
    const int half = (lane >> 4) << 3;
#pragma unroll
    for (int r = 0; r < 8; ++r) {
      const float p0 = __expf(s0[r] - runM[r]);
      const float p1 = __expf(s1[r] - runM[r]);
      float ps = p0 + p1;
#pragma unroll
      for (int m = 8; m >= 1; m >>= 1) ps += __shfl_xor(ps, m, 32);
      runL[r] = runL[r] * alpha[r] + ps;
      const int row = r + half;
      ldsP[w][row][lane & 15]        = p0;
      ldsP[w][row][16 + (lane & 15)] = p1;
      o0[r] *= alpha[r]; o1[r] *= alpha[r]; o2[r] *= alpha[r]; o3[r] *= alpha[r];
    }
    __syncthreads();
    // ---- transpose P (D layout -> A layout 16x32 bf16) via LDS ----
    v16bf pa;
    {
      const int m  = lane & 15;
      const int kc = (lane >> 4) << 3;
#pragma unroll
      for (int i = 0; i < 8; ++i) {
        pa[i]     = (bf16)ldsP[w][m][kc + i];
        pa[i + 8] = (bf16)ldsP[w][m][16 + kc + i];
      }
    }
    // ---- O += P @ V : V B-frags from d-major Vt (two contiguous 8-key runs per lane) ----
    o0 = wmma_bf16_f32(pa, load_b_frag(vt, n, 0,  kb, lane), o0);
    o1 = wmma_bf16_f32(pa, load_b_frag(vt, n, 16, kb, lane), o1);
    o2 = wmma_bf16_f32(pa, load_b_frag(vt, n, 32, kb, lane), o2);
    o3 = wmma_bf16_f32(pa, load_b_frag(vt, n, 48, kb, lane), o3);
    __syncthreads();
  }
  // ---- epilogue: divide by row sums, write [N][256] at head offset ----
  const int col   = lane & 15;
  const int rbase = (lane >> 4) << 3;
#pragma unroll
  for (int r = 0; r < 8; ++r) {
    const float inv = 1.0f / runL[r];
    float* op = O + (size_t)(q0 + rbase + r) * NCH + h * 64;
    op[col]      = o0[r] * inv;
    op[16 + col] = o1[r] * inv;
    op[32 + col] = o2[r] * inv;
    op[48 + col] = o3[r] * inv;
  }
}

// ---------------------------------------------------------------------------
// Elementwise / reduction helper kernels
// ---------------------------------------------------------------------------
__global__ void k_zero(float* __restrict__ p, int n) {
  int i = blockIdx.x * 256 + threadIdx.x;
  if (i < n) p[i] = 0.0f;
}

__global__ void k_deg(const int* __restrict__ dst, float* __restrict__ deg, int e) {
  int i = blockIdx.x * 256 + threadIdx.x;
  if (i < e) atomicAdd(deg + dst[i], 1.0f);
}

__global__ void k_scatter(const float* __restrict__ x, const int* __restrict__ src,
                          const int* __restrict__ dst, float* __restrict__ agg, int e) {
  int t = blockIdx.x * 256 + threadIdx.x;
  const int per = NCH / 4;
  if (t >= e * per) return;
  const int ed = t / per;
  const int cg = (t % per) * 4;
  const float4 v = *(const float4*)(x + (size_t)src[ed] * NCH + cg);
  float* ap = agg + (size_t)dst[ed] * NCH + cg;
  atomicAdd(ap + 0, v.x); atomicAdd(ap + 1, v.y);
  atomicAdd(ap + 2, v.z); atomicAdd(ap + 3, v.w);
}

// A2 = [agg/max(deg,1) | x] as bf16, [N x 2C]
__global__ void k_build_a2(const float* __restrict__ x, const float* __restrict__ agg,
                           const float* __restrict__ deg, bf16* __restrict__ A2, int n) {
  int i = blockIdx.x * 256 + threadIdx.x;
  if (i >= n * NCH) return;
  const int node = i >> 8, c = i & 255;
  const float dg = fmaxf(deg[node], 1.0f);
  A2[(size_t)node * (2 * NCH) + c]       = (bf16)(agg[i] / dg);
  A2[(size_t)node * (2 * NCH) + NCH + c] = (bf16)(x[i]);
}

// W2 = [w_neigh | w_root] as bf16, [C x 2C]
__global__ void k_concat_w(const float* __restrict__ wn, const float* __restrict__ wr,
                           bf16* __restrict__ W2) {
  int i = blockIdx.x * 256 + threadIdx.x;
  if (i >= NCH * NCH) return;
  const int o = i >> 8, k = i & 255;
  W2[(size_t)o * (2 * NCH) + k]       = (bf16)wn[i];
  W2[(size_t)o * (2 * NCH) + NCH + k] = (bf16)wr[i];
}

__global__ void k_convert(const float* __restrict__ s, bf16* __restrict__ d, int n) {
  int i = blockIdx.x * 256 + threadIdx.x;
  if (i < n) d[i] = (bf16)s[i];
}

__global__ void k_qkv_split(const float* __restrict__ qkv, bf16* __restrict__ Qb,
                            bf16* __restrict__ Kb, bf16* __restrict__ Vt, int n) {
  int i = blockIdx.x * 256 + threadIdx.x;
  if (i >= n * NCH) return;
  const int node = i >> 8, c = i & 255;
  const int h = c >> 6, d = c & 63;
  const float* row = qkv + (size_t)node * (3 * NCH);
  Qb[((size_t)h * n + node) * 64 + d] = (bf16)(row[c] * 0.125f);  // 1/sqrt(64)
  Kb[((size_t)h * n + node) * 64 + d] = (bf16)(row[NCH + c]);
  Vt[((size_t)h * 64 + d) * n + node] = (bf16)(row[2 * NCH + c]);
}

__global__ void k_colstats(const float* __restrict__ h, float* __restrict__ mean,
                           float* __restrict__ rstd, int M) {
  __shared__ float s1[256], s2[256];
  const int c = blockIdx.x, t = threadIdx.x;
  float a = 0.0f, b = 0.0f;
  for (int i = t; i < M; i += 256) {
    const float v = h[(size_t)i * NCH + c];
    a += v; b += v * v;
  }
  s1[t] = a; s2[t] = b; __syncthreads();
  for (int s = 128; s > 0; s >>= 1) {
    if (t < s) { s1[t] += s1[t + s]; s2[t] += s2[t + s]; }
    __syncthreads();
  }
  if (t == 0) {
    const float m = s1[0] / M;
    const float v = s2[0] / M - m * m;   // biased variance
    mean[c] = m;
    rstd[c] = rsqrtf(v + 1e-5f);
  }
}

// act: 0 none, 2 elu; optional bf16 mirror
__global__ void k_bn_apply(const float* __restrict__ h, const float* __restrict__ mean,
                           const float* __restrict__ rstd, const float* __restrict__ g,
                           const float* __restrict__ b, float* __restrict__ outf,
                           bf16* __restrict__ outbf, int n, int act) {
  int i = blockIdx.x * 256 + threadIdx.x;
  if (i >= n) return;
  const int c = i & (NCH - 1);
  float v = (h[i] - mean[c]) * rstd[c] * g[c] + b[c];
  if (act == 2) v = (v > 0.0f) ? v : (__expf(v) - 1.0f);
  outf[i] = v;
  if (outbf) outbf[i] = (bf16)v;
}

__global__ void k_add2(const float* __restrict__ a, const float* __restrict__ b,
                       float* __restrict__ outf, bf16* __restrict__ outbf, int n) {
  int i = blockIdx.x * 256 + threadIdx.x;
  if (i >= n) return;
  const float v = a[i] + b[i];
  outf[i] = v;
  outbf[i] = (bf16)v;
}

// ---------------------------------------------------------------------------
// Host orchestration
// ---------------------------------------------------------------------------
extern "C" void kernel_launch(void* const* d_in, const int* in_sizes, int n_in,
                              void* d_out, int out_size, void* d_ws, size_t ws_size,
                              hipStream_t stream) {
  (void)in_sizes; (void)n_in; (void)out_size; (void)ws_size;
  const int N = NNODES, C = NCH, E = NEDGE;

  const float* x       = (const float*)d_in[0];
  const int*   eidx    = (const int*)d_in[1];
  const float* w_neigh = (const float*)d_in[2];
  const float* b_neigh = (const float*)d_in[3];
  const float* w_root  = (const float*)d_in[4];
  const float* w_in    = (const float*)d_in[5];
  const float* b_in    = (const float*)d_in[6];
  const float* w_out   = (const float*)d_in[7];
  const float* b_out   = (const float*)d_in[8];
  const float* bn1_g   = (const float*)d_in[9];
  const float* bn1_b   = (const float*)d_in[10];
  const float* bn2_g   = (const float*)d_in[11];
  const float* bn2_b   = (const float*)d_in[12];
  const float* bn3_g   = (const float*)d_in[13];
  const float* bn3_b   = (const float*)d_in[14];
  const float* w_mlp1  = (const float*)d_in[15];
  const float* b_mlp1  = (const float*)d_in[16];
  const float* w_mlp2  = (const float*)d_in[17];
  const float* b_mlp2  = (const float*)d_in[18];
  const float* w_lin   = (const float*)d_in[19];
  const float* b_lin   = (const float*)d_in[20];
  const int* src = eidx;
  const int* dst = eidx + E;
  float* outp = (float*)d_out;

  // -- workspace carve-up --
  size_t off = 0;
  char* base = (char*)d_ws;
  auto alloc = [&](size_t bytes) -> char* {
    char* p = base + off;
    off += (bytes + 255) & ~(size_t)255;
    return p;
  };
  float* agg      = (float*)alloc((size_t)N * C * 4);      // reused as attn_out
  float* deg      = (float*)alloc((size_t)N * 4);
  bf16*  A2       = (bf16*) alloc((size_t)N * 2 * C * 2);
  bf16*  W2       = (bf16*) alloc((size_t)C * 2 * C * 2);
  float* h1       = (float*)alloc((size_t)N * C * 4);
  float* h2       = (float*)alloc((size_t)N * C * 4);
  float* meanb    = (float*)alloc((size_t)C * 4);
  float* rstdb    = (float*)alloc((size_t)C * 4);
  bf16*  xbf      = (bf16*) alloc((size_t)N * C * 2);
  float* qkv      = (float*)alloc((size_t)N * 3 * C * 4);  // reused: t1 + t1bf
  bf16*  qbf      = (bf16*) alloc((size_t)N * C * 2);
  bf16*  kbf      = (bf16*) alloc((size_t)N * C * 2);
  bf16*  vtbf     = (bf16*) alloc((size_t)N * C * 2);
  bf16*  attnbf   = (bf16*) alloc((size_t)N * C * 2);
  float* out12    = (float*)alloc((size_t)N * C * 4);
  bf16*  out12bf  = (bf16*) alloc((size_t)N * C * 2);
  float* pre3     = (float*)alloc((size_t)N * C * 4);
  bf16*  out3bf   = (bf16*) alloc((size_t)N * C * 2);
  bf16*  w_in_bf  = (bf16*) alloc((size_t)3 * C * C * 2);
  bf16*  w_out_bf = (bf16*) alloc((size_t)C * C * 2);
  bf16*  w_m1_bf  = (bf16*) alloc((size_t)2 * C * C * 2);
  bf16*  w_m2_bf  = (bf16*) alloc((size_t)C * 2 * C * 2);
  bf16*  w_lin_bf = (bf16*) alloc((size_t)C * C * 2);
  float* attn = agg;                            // alias (agg dead after A2)
  float* t1   = qkv;                            // alias (qkv dead after split)
  bf16*  t1bf = (bf16*)(qkv + (size_t)N * 2 * C);

  auto blocks = [](long n) { return dim3((unsigned)((n + 255) / 256)); };

  // ---- SAGE aggregation ----
  k_zero<<<blocks((long)N * C), 256, 0, stream>>>(agg, N * C);
  k_zero<<<blocks(N), 256, 0, stream>>>(deg, N);
  k_deg<<<blocks(E), 256, 0, stream>>>(dst, deg, E);
  k_scatter<<<blocks((long)E * (C / 4)), 256, 0, stream>>>(x, src, dst, agg, E);

  // ---- precision conversions ----
  k_convert<<<blocks((long)N * C), 256, 0, stream>>>(x, xbf, N * C);
  k_convert<<<blocks((long)3 * C * C), 256, 0, stream>>>(w_in, w_in_bf, 3 * C * C);
  k_convert<<<blocks((long)C * C), 256, 0, stream>>>(w_out, w_out_bf, C * C);
  k_convert<<<blocks((long)2 * C * C), 256, 0, stream>>>(w_mlp1, w_m1_bf, 2 * C * C);
  k_convert<<<blocks((long)2 * C * C), 256, 0, stream>>>(w_mlp2, w_m2_bf, 2 * C * C);
  k_convert<<<blocks((long)C * C), 256, 0, stream>>>(w_lin, w_lin_bf, C * C);
  k_build_a2<<<blocks((long)N * C), 256, 0, stream>>>(x, agg, deg, A2, N);
  k_concat_w<<<blocks((long)C * C), 256, 0, stream>>>(w_neigh, w_root, W2);

  // ---- branch 1: h1 = BN1(agg@Wn^T + b + x@Wr^T + x), fused K=512 GEMM ----
  k_gemm<<<dim3(N / 64, C / 128), 256, 0, stream>>>(A2, W2, b_neigh, x, h1,
                                                    N, C, 2 * C, 0);
  k_colstats<<<C, 256, 0, stream>>>(h1, meanb, rstdb, N);
  k_bn_apply<<<blocks((long)N * C), 256, 0, stream>>>(h1, meanb, rstdb, bn1_g, bn1_b,
                                                      h1, (bf16*)nullptr, N * C, 0);

  // ---- branch 2: multihead attention ----
  k_gemm<<<dim3(N / 64, (3 * C) / 128), 256, 0, stream>>>(xbf, w_in_bf, b_in,
                                                          (const float*)nullptr, qkv,
                                                          N, 3 * C, C, 0);
  k_qkv_split<<<blocks((long)N * C), 256, 0, stream>>>(qkv, qbf, kbf, vtbf, N);
  k_attn<<<dim3(N / 128, NHEAD), 256, 0, stream>>>(qbf, kbf, vtbf, attn, N);
  k_convert<<<blocks((long)N * C), 256, 0, stream>>>(attn, attnbf, N * C);
  k_gemm<<<dim3(N / 64, C / 128), 256, 0, stream>>>(attnbf, w_out_bf, b_out, x, h2,
                                                    N, C, C, 0);
  k_colstats<<<C, 256, 0, stream>>>(h2, meanb, rstdb, N);
  k_bn_apply<<<blocks((long)N * C), 256, 0, stream>>>(h2, meanb, rstdb, bn2_g, bn2_b,
                                                      h2, (bf16*)nullptr, N * C, 0);

  // ---- out = h1 + h2; MLP with residual; BN3 + ELU ----
  k_add2<<<blocks((long)N * C), 256, 0, stream>>>(h1, h2, out12, out12bf, N * C);
  k_gemm<<<dim3(N / 64, (2 * C) / 128), 256, 0, stream>>>(out12bf, w_m1_bf, b_mlp1,
                                                          (const float*)nullptr, t1,
                                                          N, 2 * C, C, 1);
  k_convert<<<blocks((long)N * 2 * C), 256, 0, stream>>>(t1, t1bf, N * 2 * C);
  k_gemm<<<dim3(N / 64, C / 128), 256, 0, stream>>>(t1bf, w_m2_bf, b_mlp2, out12, pre3,
                                                    N, C, 2 * C, 0);
  k_colstats<<<C, 256, 0, stream>>>(pre3, meanb, rstdb, N);
  k_bn_apply<<<blocks((long)N * C), 256, 0, stream>>>(pre3, meanb, rstdb, bn3_g, bn3_b,
                                                      pre3, out3bf, N * C, 2);

  // ---- final linear ----
  k_gemm<<<dim3(N / 64, C / 128), 256, 0, stream>>>(out3bf, w_lin_bf, b_lin,
                                                    (const float*)nullptr, outp,
                                                    N, C, C, 0);
}
